// RegressionModel_18794776887871
// MI455X (gfx1250) — matrix operand order repaired and checked by reading, not compiled, not run on hardware
//
#include <hip/hip_runtime.h>
#include <hip/hip_bf16.h>
#include <math.h>

// ---------------------------------------------------------------------------
// CDNA5 (gfx1250) 12-layer sliding-window transformer.
// wave32; v_wmma_f32_16x16x32_f16 for all GEMM/attention math; f16
// weights/activations feed the matrix pipes; f32 residual stream for LN.
// LDS tiles laid out so every WMMA fragment load is 2x ds_load_b128.
// Straight-through tiles use GLOBAL_LOAD_ASYNC_TO_LDS (ASYNCcnt path).
// ---------------------------------------------------------------------------

typedef __attribute__((ext_vector_type(16))) _Float16 v16h;
typedef __attribute__((ext_vector_type(8)))  float    v8f;
typedef __attribute__((ext_vector_type(4)))  int      i4;

constexpr int BATCH = 2;
constexpr int SEQ   = 4096;
constexpr int DM    = 768;
constexpr int NH    = 12;
constexpr int DHD   = 64;
constexpr int FF    = 3072;
constexpr int WIN   = 256;
constexpr int NL    = 12;
constexpr int MTOK  = BATCH * SEQ;   // 8192 tokens

// -------- async global->LDS copy (guarded; falls back to sync b128) --------
#ifdef __HIP_DEVICE_COMPILE__
#if __has_builtin(__builtin_amdgcn_global_load_async_to_lds_b128) && \
    __has_builtin(__builtin_amdgcn_s_wait_asynccnt)
#define HAVE_ASYNC_LDS 1
#endif
#endif
#ifndef HAVE_ASYNC_LDS
#define HAVE_ASYNC_LDS 0
#endif

#if HAVE_ASYNC_LDS
typedef __attribute__((address_space(1))) i4 gi4;   // global int4
typedef __attribute__((address_space(3))) i4 li4;   // LDS int4
__device__ inline void async_copy16(void* lds_dst, const void* g_src) {
    __builtin_amdgcn_global_load_async_to_lds_b128(
        (gi4*)(unsigned long long)g_src,
        (li4*)(unsigned)(unsigned long long)lds_dst,
        0, 0);
}
__device__ inline void wait_async() { __builtin_amdgcn_s_wait_asynccnt(0); }
#else
__device__ inline void wait_async() {}
#endif

// ---------------------------------------------------------------------------
// helpers
// ---------------------------------------------------------------------------
__device__ inline v8f vzero8() {
    v8f z = {0.f, 0.f, 0.f, 0.f, 0.f, 0.f, 0.f, 0.f};
    return z;
}

__device__ inline v8f wmma16(v16h a, v16h b, v8f c) {
    return __builtin_amdgcn_wmma_f32_16x16x32_f16(
        /*neg_a=*/false, a, /*neg_b=*/false, b,
        /*c_mod=*/(short)0, c, /*reuse_a=*/false, /*reuse_b=*/false);
}

__device__ inline float fast_rcp(float x) {
#if __has_builtin(__builtin_amdgcn_rcpf)
    return __builtin_amdgcn_rcpf(x);
#else
    return 1.f / x;
#endif
}

// A fragment: 16x32 (MxK) row-major, leading dim ld (ISA 7.12.2 layout).
__device__ inline v16h load_a_frag(const _Float16* base, int ld) {
    const int lane = threadIdx.x & 31;
    const int m    = lane & 15;
    const int hf   = lane >> 4;
    const _Float16* row = base + (size_t)m * ld;
    v16h f;
#pragma unroll
    for (int v = 0; v < 8; ++v) {
        int kb = ((v & 4) ? 16 : 0) + hf * 8 + (v & 3) * 2;
        f[2 * v]     = row[kb];
        f[2 * v + 1] = row[kb + 1];
    }
    return f;
}

// B fragment from K-transposed storage: logical B(k,n) = base[n*ld + k].
__device__ inline v16h load_bT_frag(const _Float16* base, int ld) {
    const int lane = threadIdx.x & 31;
    const int n    = lane & 15;
    const int kh   = (lane >> 4) * 8;
    const _Float16* row = base + (size_t)n * ld + kh;
    v16h f;
#pragma unroll
    for (int v = 0; v < 8; ++v) {
        int kb = ((v & 4) ? 16 : 0) + (v & 3) * 2;
        f[2 * v]     = row[kb];
        f[2 * v + 1] = row[kb + 1];
    }
    return f;
}

__device__ inline float hmax16(float v) {
#pragma unroll
    for (int m = 1; m < 16; m <<= 1) v = fmaxf(v, __shfl_xor(v, m, 32));
    return v;
}
__device__ inline float hsum16(float v) {
#pragma unroll
    for (int m = 1; m < 16; m <<= 1) v += __shfl_xor(v, m, 32);
    return v;
}
__device__ inline float wsum32(float v) {
#pragma unroll
    for (int m = 1; m < 32; m <<= 1) v += __shfl_xor(v, m, 32);
    return v;
}

__device__ inline float block_sum(float v, float* red) {
    const int t = threadIdx.x, lane = t & 31, w = t >> 5;
    v = wsum32(v);
    if (lane == 0) red[w] = v;
    __syncthreads();
    float r = (t < 8) ? red[t] : 0.f;
    r += __shfl_xor(r, 1, 32);
    r += __shfl_xor(r, 2, 32);
    r += __shfl_xor(r, 4, 32);
    if (t == 0) red[0] = r;
    __syncthreads();
    r = red[0];
    __syncthreads();
    return r;
}

// branch-free exact GELU: erf via Abramowitz-Stegun 7.1.26 (|err| < 1.5e-7)
__device__ inline float gelu_exact(float x) {
    float ax   = fabsf(x) * 0.70710678118f;
    float t    = fast_rcp(1.f + 0.3275911f * ax);
    float poly = t * (0.254829592f +
                 t * (-0.284496736f +
                 t * (1.421413741f +
                 t * (-1.453152027f + t * 1.061405429f))));
    float erf_ax = 1.f - poly * __expf(-ax * ax);
    float erfv   = copysignf(erf_ax, x);
    return 0.5f * x * (1.f + erfv);
}

// ---------------------------------------------------------------------------
// fp32 -> f16 weight conversion
// ---------------------------------------------------------------------------
__global__ void cvt_kernel(const float* __restrict__ s, _Float16* __restrict__ d,
                           size_t n) {
    size_t i      = (size_t)blockIdx.x * 256 + threadIdx.x;
    size_t stride = (size_t)gridDim.x * 256;
    for (; i < n; i += stride) d[i] = (_Float16)s[i];
}

// ---------------------------------------------------------------------------
// Embedding gather + LayerNorm (one block per token)
// ---------------------------------------------------------------------------
__global__ void embed_kernel(const int* __restrict__ ids,
                             const float* __restrict__ we,
                             const float* __restrict__ pe,
                             const float* __restrict__ te,
                             const float* __restrict__ g,
                             const float* __restrict__ bb,
                             float* __restrict__ h32,
                             _Float16* __restrict__ h16) {
    __shared__ float red[8];
    const int tok = blockIdx.x;
    const int t   = threadIdx.x;
    const int id  = ids[tok];
    const int pos = (tok % SEQ) + 2;

    float vals[3];
    float s = 0.f;
#pragma unroll
    for (int i = 0; i < 3; ++i) {
        int c   = t + i * 256;
        float v = we[(size_t)id * DM + c] + pe[(size_t)pos * DM + c] + te[c];
        vals[i] = v;
        s += v;
    }
    float mu = block_sum(s, red) * (1.f / DM);
    s = 0.f;
#pragma unroll
    for (int i = 0; i < 3; ++i) {
        float d = vals[i] - mu;
        s += d * d;
    }
    float var  = block_sum(s, red) * (1.f / DM);
    float rstd = rsqrtf(var + 1e-5f);
#pragma unroll
    for (int i = 0; i < 3; ++i) {
        int c   = t + i * 256;
        float y = (vals[i] - mu) * rstd * g[c] + bb[c];
        h32[(size_t)tok * DM + c] = y;
        h16[(size_t)tok * DM + c] = (_Float16)y;
    }
}

// ---------------------------------------------------------------------------
// Residual add + LayerNorm (res and h32 alias; threads own their elements)
// ---------------------------------------------------------------------------
__global__ void add_ln_kernel(const float* res, const float* __restrict__ delta,
                              const float* __restrict__ g,
                              const float* __restrict__ bb, float* h32,
                              _Float16* __restrict__ h16) {
    __shared__ float red[8];
    const int tok = blockIdx.x;
    const int t   = threadIdx.x;
    const size_t base = (size_t)tok * DM;

    float vals[3];
    float s = 0.f;
#pragma unroll
    for (int i = 0; i < 3; ++i) {
        int c   = t + i * 256;
        float v = res[base + c] + delta[base + c];
        vals[i] = v;
        s += v;
    }
    float mu = block_sum(s, red) * (1.f / DM);
    s = 0.f;
#pragma unroll
    for (int i = 0; i < 3; ++i) {
        float d = vals[i] - mu;
        s += d * d;
    }
    float var  = block_sum(s, red) * (1.f / DM);
    float rstd = rsqrtf(var + 1e-5f);
#pragma unroll
    for (int i = 0; i < 3; ++i) {
        int c   = t + i * 256;
        float y = (vals[i] - mu) * rstd * g[c] + bb[c];
        h32[base + c] = y;
        h16[base + c] = (_Float16)y;
    }
}

// ---------------------------------------------------------------------------
// Tiled WMMA GEMM: C[M,N] = act((A[M,K] @ W[K,N] + bias) * scale)
// Block tile 128x128, BK=64 (32KB LDS). W tile stored K-transposed so every
// fragment load is 2x ds_load_b128. Next K-tile prefetched near (loc=3).
// A tile copied global->LDS asynchronously when available.
// ---------------------------------------------------------------------------
__global__ __launch_bounds__(256) void gemm_kernel(
    const _Float16* __restrict__ A, const _Float16* __restrict__ Wt,
    const float* __restrict__ bias, float* __restrict__ out32,
    _Float16* __restrict__ out16, int M, int N, int K, int act, float scale) {
    __shared__ __align__(16) _Float16 As[128 * 64];    // [m][k]
    __shared__ __align__(16) _Float16 BsT[128 * 64];   // [n][k]

    const int t   = threadIdx.x;
    const int wid = t >> 5;
    const int rb  = wid >> 2;  // 0..1 : 64-row block
    const int cb  = wid & 3;   // 0..3 : 32-col block
    const int n0  = blockIdx.x * 128;
    const int m0  = blockIdx.y * 128;
    const int nl  = t & 15;
    const int hf  = (t & 31) >> 4;

    v8f acc[4][2];
#pragma unroll
    for (int mi = 0; mi < 4; ++mi)
#pragma unroll
        for (int ni = 0; ni < 2; ++ni) acc[mi][ni] = vzero8();

    const int arow = t >> 1, aseg = t & 1;   // A loader: 2 thr/row, 64B each
    const int krow = t >> 2, nseg = t & 3;   // W loader: 4 thr/row, 64B each

    for (int k0 = 0; k0 < K; k0 += 64) {
        // prefetch next K-tile into near caches while computing on this one
        if (k0 + 64 < K) {
            __builtin_prefetch(A + (size_t)(m0 + arow) * K + (k0 + 64) + aseg * 32, 0, 3);
            __builtin_prefetch(Wt + (size_t)(k0 + 64 + krow) * N + n0 + nseg * 32, 0, 3);
        }
        {   // A tile: 128 rows x 64 halves, row-major (straight copy)
            const float4* src = reinterpret_cast<const float4*>(
                A + (size_t)(m0 + arow) * K + k0 + aseg * 32);
            float4* dst = reinterpret_cast<float4*>(As + arow * 64 + aseg * 32);
#if HAVE_ASYNC_LDS
            async_copy16(dst + 0, src + 0);
            async_copy16(dst + 1, src + 1);
            async_copy16(dst + 2, src + 2);
            async_copy16(dst + 3, src + 3);
#else
            float4 t0 = src[0], t1 = src[1], t2 = src[2], t3 = src[3];
            dst[0] = t0; dst[1] = t1; dst[2] = t2; dst[3] = t3;
#endif
        }
        {   // W tile: 64 k-rows x 128 n-cols, scattered into BsT[n][k]
            const float4* src = reinterpret_cast<const float4*>(
                Wt + (size_t)(k0 + krow) * N + n0 + nseg * 32);
            __align__(16) _Float16 tmp[32];
            float4 w0 = src[0], w1 = src[1], w2 = src[2], w3 = src[3];
            reinterpret_cast<float4*>(tmp)[0] = w0;
            reinterpret_cast<float4*>(tmp)[1] = w1;
            reinterpret_cast<float4*>(tmp)[2] = w2;
            reinterpret_cast<float4*>(tmp)[3] = w3;
#pragma unroll
            for (int i = 0; i < 32; ++i)
                BsT[(nseg * 32 + i) * 64 + krow] = tmp[i];
        }
        wait_async();
        __syncthreads();

#pragma unroll
        for (int kk = 0; kk < 64; kk += 32) {
            v16h af[4], bf2[2];
#pragma unroll
            for (int mi = 0; mi < 4; ++mi)
                af[mi] = load_a_frag(As + (rb * 64 + mi * 16) * 64 + kk, 64);
#pragma unroll
            for (int ni = 0; ni < 2; ++ni)
                bf2[ni] = load_bT_frag(BsT + (cb * 32 + ni * 16) * 64 + kk, 64);
#pragma unroll
            for (int mi = 0; mi < 4; ++mi)
#pragma unroll
                for (int ni = 0; ni < 2; ++ni)
                    acc[mi][ni] = wmma16(af[mi], bf2[ni], acc[mi][ni]);
        }
        __syncthreads();
    }

    // Epilogue: bias, scale, activation, stores
#pragma unroll
    for (int mi = 0; mi < 4; ++mi) {
#pragma unroll
        for (int ni = 0; ni < 2; ++ni) {
#pragma unroll
            for (int r = 0; r < 8; ++r) {
                int row = m0 + rb * 64 + mi * 16 + r + 8 * hf;
                int col = n0 + cb * 32 + ni * 16 + nl;
                float v = (acc[mi][ni][r] + bias[col]) * scale;
                if (act == 1) v = gelu_exact(v);
                size_t idx = (size_t)row * N + col;
                if (out32) out32[idx] = v;
                if (out16) out16[idx] = (_Float16)v;
            }
        }
    }
}

// ---------------------------------------------------------------------------
// Sliding-window flash attention.
// One block per (batch, head, 128-query strip); 8 waves x 16 queries.
// 20 key tiles of 32 cover [qbase-256, qbase+384); band |key-q| <= 256.
// OOB keys are clamped for the load and neutralized by the -1e9 mask / p=0,
// so the K-tile copy is unconditional (async-copy friendly).
// ---------------------------------------------------------------------------
__global__ __launch_bounds__(256) void attn_kernel(
    const _Float16* __restrict__ q16, const _Float16* __restrict__ k16,
    const _Float16* __restrict__ v16g, const int* __restrict__ amask,
    _Float16* __restrict__ a16) {
    __shared__ __align__(16) _Float16 Kt[32 * DHD];    // [key][dh]
    __shared__ __align__(16) _Float16 VtT[DHD * 32];   // [dh][key]
    __shared__ __align__(16) _Float16 Pst[8 * 16 * 32];
    __shared__ int kvalid[32];

    const int t    = threadIdx.x;
    const int lane = t & 31;
    const int wid  = t >> 5;
    const int nl   = lane & 15;
    const int hf   = lane >> 4;

    const int nqb   = SEQ / 128;
    const int qb    = blockIdx.x % nqb;
    const int head  = (blockIdx.x / nqb) % NH;
    const int b     = blockIdx.x / (nqb * NH);
    const int qbase = qb * 128;
    const int q0    = qbase + wid * 16;
    const size_t rowoff = (size_t)b * SEQ;

    const _Float16* qptr = q16 + (rowoff + q0) * DM + head * DHD;
    v16h qf[2];
    qf[0] = load_a_frag(qptr, DM);
    qf[1] = load_a_frag(qptr + 32, DM);

    v8f oacc[4];
#pragma unroll
    for (int j = 0; j < 4; ++j) oacc[j] = vzero8();
    float mrun[8], lrun[8];
#pragma unroll
    for (int r = 0; r < 8; ++r) { mrun[r] = -1e30f; lrun[r] = 0.f; }

    _Float16* Pw = Pst + wid * (16 * 32);
    const int krow = t >> 3, kseg = t & 7;   // K/V loaders: 8 thr/key

    for (int tt = 0; tt < 20; ++tt) {
        const int ks = qbase - WIN + tt * 32;
        {   // cooperative K/V tile load: 32 keys x 64 dh (clamped addresses)
            int key  = ks + krow;
            int keyc = key < 0 ? 0 : (key >= SEQ ? SEQ - 1 : key);
            size_t off = (rowoff + keyc) * DM + head * DHD + kseg * 8;
#if HAVE_ASYNC_LDS
            async_copy16(Kt + krow * DHD + kseg * 8, k16 + off);
#else
            *reinterpret_cast<float4*>(Kt + krow * DHD + kseg * 8) =
                *reinterpret_cast<const float4*>(k16 + off);
#endif
            float4 vv = *reinterpret_cast<const float4*>(v16g + off);
            __align__(16) _Float16 vtmp[8];
            *reinterpret_cast<float4*>(vtmp) = vv;
#pragma unroll
            for (int i = 0; i < 8; ++i)
                VtT[(kseg * 8 + i) * 32 + krow] = vtmp[i];
            if (t < 32) {
                int key2 = ks + t;
                kvalid[t] =
                    (key2 >= 0 && key2 < SEQ && amask[b * SEQ + key2] != 0) ? 1 : 0;
            }
        }
        wait_async();
        __syncthreads();

        // scores: S(16q x 32k) = Q(16x64) * K^T(64x32)
        v8f sc[2];
#pragma unroll
        for (int h2 = 0; h2 < 2; ++h2) {
            v8f s = vzero8();
            s = wmma16(qf[0], load_bT_frag(Kt + (h2 * 16) * DHD, DHD), s);
            s = wmma16(qf[1], load_bT_frag(Kt + (h2 * 16) * DHD + 32, DHD), s);
            sc[h2] = s;
        }

        // band + key mask
#pragma unroll
        for (int h2 = 0; h2 < 2; ++h2) {
#pragma unroll
            for (int r = 0; r < 8; ++r) {
                int key  = ks + h2 * 16 + nl;
                int qrow = q0 + r + 8 * hf;
                int dk   = key - qrow;
                bool ok  = (kvalid[h2 * 16 + nl] != 0) && dk <= WIN && dk >= -WIN;
                if (!ok) sc[h2][r] = -1e9f;
            }
        }

        // online softmax
        float mnew[8], scal[8];
#pragma unroll
        for (int r = 0; r < 8; ++r) {
            float tm = hmax16(fmaxf(sc[0][r], sc[1][r]));
            mnew[r]  = fmaxf(mrun[r], tm);
            scal[r]  = __expf(mrun[r] - mnew[r]);
            mrun[r]  = mnew[r];
        }
#pragma unroll
        for (int r = 0; r < 8; ++r) {
            float p0 = (sc[0][r] < -1e8f) ? 0.f : __expf(sc[0][r] - mnew[r]);
            float p1 = (sc[1][r] < -1e8f) ? 0.f : __expf(sc[1][r] - mnew[r]);
            sc[0][r] = p0;
            sc[1][r] = p1;
            lrun[r]  = lrun[r] * scal[r] + hsum16(p0 + p1);
        }
#pragma unroll
        for (int j = 0; j < 4; ++j)
#pragma unroll
            for (int r = 0; r < 8; ++r) oacc[j][r] *= scal[r];

        // restage P into A-fragment layout via per-wave LDS
#pragma unroll
        for (int h2 = 0; h2 < 2; ++h2)
#pragma unroll
            for (int r = 0; r < 8; ++r)
                Pw[(r + 8 * hf) * 32 + h2 * 16 + nl] = (_Float16)sc[h2][r];

        v16h pf = load_a_frag(Pw, 32);
#pragma unroll
        for (int j = 0; j < 4; ++j)
            oacc[j] = wmma16(pf, load_bT_frag(VtT + (j * 16) * 32, 32), oacc[j]);
        __syncthreads();
    }

    // normalize + store (f16 feeds Wo projection)
#pragma unroll
    for (int r = 0; r < 8; ++r) {
        float rinv = fast_rcp(lrun[r]);
        int m      = r + 8 * hf;
#pragma unroll
        for (int j = 0; j < 4; ++j) {
            float ov = oacc[j][r] * rinv;
            a16[(rowoff + q0 + m) * DM + head * DHD + j * 16 + nl] = (_Float16)ov;
        }
    }
}

// ---------------------------------------------------------------------------
// Regression head: out[b] = relu(h[b,0] @ Wr1 + br1) @ Wr2 + br2
// ---------------------------------------------------------------------------
__global__ void head_kernel(const float* __restrict__ h32,
                            const float* __restrict__ Wr1,
                            const float* __restrict__ br1,
                            const float* __restrict__ Wr2,
                            const float* __restrict__ br2,
                            float* __restrict__ out) {
    __shared__ float pooled[DM];
    __shared__ float hid[512];
    __shared__ float red[8];
    const int b = blockIdx.x;
    const int t = threadIdx.x;

    for (int c = t; c < DM; c += 256) pooled[c] = h32[(size_t)b * SEQ * DM + c];
    __syncthreads();
    for (int j = t; j < 512; j += 256) {
        float s = br1[j];
        for (int i = 0; i < DM; ++i) s += pooled[i] * Wr1[(size_t)i * 512 + j];
        hid[j] = fmaxf(s, 0.f);
    }
    __syncthreads();
    float s = 0.f;
    for (int j = t; j < 512; j += 256) s += hid[j] * Wr2[j];
    float tot = block_sum(s, red);
    if (t == 0) out[b] = tot + br2[0];
}

// ---------------------------------------------------------------------------
// Host orchestration
// ---------------------------------------------------------------------------
extern "C" void kernel_launch(void* const* d_in, const int* in_sizes, int n_in,
                              void* d_out, int out_size, void* d_ws,
                              size_t ws_size, hipStream_t stream) {
    (void)in_sizes; (void)n_in; (void)out_size; (void)ws_size;

    const int*   ids   = (const int*)d_in[0];
    const int*   amask = (const int*)d_in[1];
    const float* we    = (const float*)d_in[2];
    const float* pe    = (const float*)d_in[3];
    const float* te    = (const float*)d_in[4];
    const float* elnw  = (const float*)d_in[5];
    const float* elnb  = (const float*)d_in[6];
    const float* Wq    = (const float*)d_in[7];
    const float* bq    = (const float*)d_in[8];
    const float* Wk    = (const float*)d_in[9];
    const float* bk    = (const float*)d_in[10];
    const float* Wv    = (const float*)d_in[11];
    const float* bv    = (const float*)d_in[12];
    const float* Wo    = (const float*)d_in[13];
    const float* bo    = (const float*)d_in[14];
    const float* ln1w  = (const float*)d_in[15];
    const float* ln1b  = (const float*)d_in[16];
    const float* Wi    = (const float*)d_in[17];
    const float* bi    = (const float*)d_in[18];
    const float* Wf    = (const float*)d_in[19];
    const float* bfp   = (const float*)d_in[20];
    const float* ln2w  = (const float*)d_in[21];
    const float* ln2b  = (const float*)d_in[22];
    const float* Wr1   = (const float*)d_in[23];
    const float* br1   = (const float*)d_in[24];
    const float* Wr2   = (const float*)d_in[25];
    const float* br2   = (const float*)d_in[26];
    float* out = (float*)d_out;

    char* p = (char*)d_ws;
    auto alloc = [&](size_t bytes) -> char* {
        char* r = p;
        p += (bytes + 255) & ~(size_t)255;
        return r;
    };
    const size_t nDD = (size_t)NL * DM * DM;
    const size_t nDF = (size_t)NL * DM * FF;
    const size_t MD  = (size_t)MTOK * DM;
    const size_t MF  = (size_t)MTOK * FF;

    _Float16* wq16 = (_Float16*)alloc(nDD * 2);
    _Float16* wk16 = (_Float16*)alloc(nDD * 2);
    _Float16* wv16 = (_Float16*)alloc(nDD * 2);
    _Float16* wo16 = (_Float16*)alloc(nDD * 2);
    _Float16* wi16 = (_Float16*)alloc(nDF * 2);
    _Float16* wf16 = (_Float16*)alloc(nDF * 2);
    float*    h32   = (float*)alloc(MD * 4);
    float*    tmp32 = (float*)alloc(MD * 4);
    _Float16* h16   = (_Float16*)alloc(MD * 2);
    _Float16* q16   = (_Float16*)alloc(MD * 2);
    _Float16* k16   = (_Float16*)alloc(MD * 2);
    _Float16* v16b  = (_Float16*)alloc(MD * 2);
    _Float16* a16   = (_Float16*)alloc(MD * 2);
    _Float16* ff16  = (_Float16*)alloc(MF * 2);

    cvt_kernel<<<2048, 256, 0, stream>>>(Wq, wq16, nDD);
    cvt_kernel<<<2048, 256, 0, stream>>>(Wk, wk16, nDD);
    cvt_kernel<<<2048, 256, 0, stream>>>(Wv, wv16, nDD);
    cvt_kernel<<<2048, 256, 0, stream>>>(Wo, wo16, nDD);
    cvt_kernel<<<4096, 256, 0, stream>>>(Wi, wi16, nDF);
    cvt_kernel<<<4096, 256, 0, stream>>>(Wf, wf16, nDF);

    embed_kernel<<<MTOK, 256, 0, stream>>>(ids, we, pe, te, elnw, elnb, h32, h16);

    const dim3 gP(DM / 128, MTOK / 128);   // 6 x 64
    const dim3 gI(FF / 128, MTOK / 128);   // 24 x 64
    const int attnBlocks = BATCH * NH * (SEQ / 128);

    for (int l = 0; l < NL; ++l) {
        const size_t wDD = (size_t)l * DM * DM;
        const size_t wDF = (size_t)l * DM * FF;
        gemm_kernel<<<gP, 256, 0, stream>>>(h16, wq16 + wDD, bq + l * DM,
                                            nullptr, q16, MTOK, DM, DM, 0, 0.125f);
        gemm_kernel<<<gP, 256, 0, stream>>>(h16, wk16 + wDD, bk + l * DM,
                                            nullptr, k16, MTOK, DM, DM, 0, 1.f);
        gemm_kernel<<<gP, 256, 0, stream>>>(h16, wv16 + wDD, bv + l * DM,
                                            nullptr, v16b, MTOK, DM, DM, 0, 1.f);
        attn_kernel<<<attnBlocks, 256, 0, stream>>>(q16, k16, v16b, amask, a16);
        gemm_kernel<<<gP, 256, 0, stream>>>(a16, wo16 + wDD, bo + l * DM,
                                            tmp32, nullptr, MTOK, DM, DM, 0, 1.f);
        add_ln_kernel<<<MTOK, 256, 0, stream>>>(h32, tmp32, ln1w + l * DM,
                                                ln1b + l * DM, h32, h16);
        gemm_kernel<<<gI, 256, 0, stream>>>(h16, wi16 + wDF, bi + (size_t)l * FF,
                                            nullptr, ff16, MTOK, FF, DM, 1, 1.f);
        gemm_kernel<<<gP, 256, 0, stream>>>(ff16, wf16 + wDF, bfp + l * DM,
                                            tmp32, nullptr, MTOK, DM, FF, 0, 1.f);
        add_ln_kernel<<<MTOK, 256, 0, stream>>>(h32, tmp32, ln2w + l * DM,
                                                ln2b + l * DM, h32, h16);
    }

    head_kernel<<<BATCH, 256, 0, stream>>>(h32, Wr1, br1, Wr2, br2, out);
}
